// Decoder_28235115004611
// MI455X (gfx1250) — matrix operand order repaired and checked by reading, not compiled, not run on hardware
//
#include <hip/hip_runtime.h>

typedef unsigned short u16;
typedef __attribute__((ext_vector_type(16))) __bf16 v16bf;
typedef __attribute__((ext_vector_type(8)))  float  v8f;
typedef int gv4i __attribute__((vector_size(16)));   // matches builtin's V4i param
typedef __attribute__((address_space(1))) gv4i* gas1_v4i;
typedef __attribute__((address_space(3))) gv4i* gas3_v4i;

#define HID   768
#define FOURH 3072
#define BATCH 64
#define SEQT  1024
#define INP   512
#define OUTD  100
#define ROWS  (SEQT*BATCH)   // 65536 timestep-rows
#define NH    8              // hidden units per scan workgroup
#define SCAN_WGS (HID/NH)    // 96 persistent workgroups
#define SLABSTR (HID+8)      // padded LDS row stride (u16) -> rotates banks

#if defined(__gfx1250__) && __has_builtin(__builtin_amdgcn_global_load_async_to_lds_b128) && __has_builtin(__builtin_amdgcn_s_wait_asynccnt)
#define USE_ASYNC_LDS 1
#else
#define USE_ASYNC_LDS 0
#endif

// ---------- helpers ----------
__device__ __forceinline__ u16 f2bf(float f){
  unsigned u = __float_as_uint(f);
  unsigned r = u + 0x7FFFu + ((u >> 16) & 1u);   // round-to-nearest-even
  return (u16)(r >> 16);
}
__device__ __forceinline__ float bf2f(u16 h){ return __uint_as_float(((unsigned)h) << 16); }

// A fragment (16x32 bf16, row-major source): lane = row, documented K interleave
__device__ __forceinline__ v16bf load_a_frag(const u16* row, int k0, int hf){
  v16bf v;
  __builtin_memcpy(&v,                                  row + k0 + 8*hf,      16);
  __builtin_memcpy(reinterpret_cast<char*>(&v) + 16,    row + k0 + 16 + 8*hf, 16);
  return v;
}
// B fragment (32x16 bf16): lane = column n, same K interleave as A (B[k][n] = W[n][k])
__device__ __forceinline__ v16bf load_b_frag(const u16* row, int k0, int hf){
  v16bf v;
  __builtin_memcpy(&v,                                  row + k0 + 8*hf,      16);
  __builtin_memcpy(reinterpret_cast<char*>(&v) + 16,    row + k0 + 16 + 8*hf, 16);
  return v;
}
__device__ __forceinline__ v8f wmma_bf16(v16bf a, v16bf b, v8f c){
  return __builtin_amdgcn_wmma_f32_16x16x32_bf16(false, a, false, b, (short)0, c, false, false);
}
__device__ __forceinline__ float sigm(float x){
  x = fminf(15.f, fmaxf(-15.f, x));
  return 1.f / (1.f + __expf(-x));
}
__device__ __forceinline__ float tanh_f(float x){
  x = fminf(15.f, fmaxf(-15.f, x));
  float e = __expf(2.f * x);
  return (e - 1.f) / (e + 1.f);
}

// ---------- conversion kernels ----------
// x [B,T,In] f32 -> [T,B,In] bf16 (transpose so GEMM rows are t*B+b)
__global__ __launch_bounds__(256) void k_cvt_x(const float* __restrict__ x, u16* __restrict__ o){
  size_t e = (size_t)blockIdx.x * 256 + threadIdx.x;     // over ROWS*INP
  int    c   = (int)(e & (INP - 1));
  size_t rtb = e >> 9;                                   // INP = 512 = 2^9
  int t = (int)(rtb >> 6);                               // BATCH = 64 = 2^6
  int b = (int)(rtb & 63);
  o[e] = f2bf(x[((size_t)b * SEQT + t) * INP + c]);
}
__global__ __launch_bounds__(256) void k_cvt(const float* __restrict__ s, u16* __restrict__ d, int n){
  int i = blockIdx.x * 256 + threadIdx.x;
  if (i < n) d[i] = f2bf(s[i]);
}
// Wp [100,768] -> zero-padded [112,768] bf16
__global__ __launch_bounds__(256) void k_cvt_wp(const float* __restrict__ s, u16* __restrict__ d){
  int i = blockIdx.x * 256 + threadIdx.x;                // over 112*768
  int r = i / HID, c = i - r * HID;
  d[i] = (r < OUTD) ? f2bf(s[r * HID + c]) : (u16)0;
}
__global__ __launch_bounds__(256) void k_zero(unsigned* __restrict__ p, int n){
  int i = blockIdx.x * 256 + threadIdx.x;
  if (i < n) p[i] = 0u;
}

// ---------- big input GEMM: pre[m,n] = sum_k A[m,k]*W[n,k] + bias[n] ----------
// A: bf16 [ROWS x K] ([T,B,*] order), W: bf16 [FOURH x K], out: f32 [ROWS x FOURH]
// WG tile 64x256; wave tile 32x64 (2x4 WMMA tiles): 6 frag loads -> 8 WMMAs/K-step.
__global__ __launch_bounds__(256) void k_gemm_pre(const u16* __restrict__ A,
                                                  const u16* __restrict__ W,
                                                  const float* __restrict__ bias,
                                                  float* __restrict__ out, int K){
  const int tid  = threadIdx.x;
  const int wave = tid >> 5, lane = tid & 31;
  const int lm = lane & 15, hf = lane >> 4;
  const int wm = wave & 1, wn = wave >> 1;               // 2 x 4 wave grid
  const size_t mbase = (size_t)blockIdx.x * 64 + wm * 32;
  const int    nbase = blockIdx.y * 256 + wn * 64;

  v8f acc[2][4];
  #pragma unroll
  for (int i = 0; i < 2; ++i)
    #pragma unroll
    for (int j = 0; j < 4; ++j)
      acc[i][j] = (v8f){0.f,0.f,0.f,0.f,0.f,0.f,0.f,0.f};

  const u16* arow0 = A + (mbase + lm) * K;
  const u16* arow1 = A + (mbase + 16 + lm) * K;
  const u16* brow[4];
  #pragma unroll
  for (int j = 0; j < 4; ++j)
    brow[j] = W + (size_t)(nbase + j * 16 + lm) * K;

  for (int k = 0; k < K; k += 32){
    v16bf a0 = load_a_frag(arow0, k, hf);
    v16bf a1 = load_a_frag(arow1, k, hf);
    #pragma unroll
    for (int j = 0; j < 4; ++j){
      v16bf b = load_b_frag(brow[j], k, hf);
      acc[0][j] = wmma_bf16(a0, b, acc[0][j]);
      acc[1][j] = wmma_bf16(a1, b, acc[1][j]);
    }
  }
  #pragma unroll
  for (int i = 0; i < 2; ++i)
    #pragma unroll
    for (int j = 0; j < 4; ++j)
      #pragma unroll
      for (int r = 0; r < 8; ++r){
        size_t m = mbase + i * 16 + r + 8 * hf;          // C layout: M = r + 8*(lane/16)
        int    n = nbase + j * 16 + lm;                  //           N = lane%16
        out[m * FOURH + n] = acc[i][j][r] + bias[n];
      }
}

// ---------- persistent recurrent scan ----------
// Each of 96 WGs owns NH=8 hidden units (gate cols n0+{0,H,2H,3H}..), stages its
// 32 Whh rows into LDS once (async-to-LDS when available), then loops t=0..T-1:
// gates = h_t*Whh^T + pre[t], LSTM cell, writes h_{t+1} (double-buffered) +
// out_seq (+residual), device-wide step barrier. Prefetches next step's pre.
__global__ __launch_bounds__(256) void k_scan(const float* __restrict__ pre,
                                              const u16* __restrict__ Whh,
                                              const u16* __restrict__ inseq,
                                              u16* __restrict__ outseq,
                                              u16* __restrict__ hbuf,
                                              unsigned* __restrict__ cnt,
                                              int useRes){
  __shared__ __align__(16) u16 slab[4 * NH * SLABSTR];   // 32 rows * 776 u16 = 48.5 KB
  __shared__ float gatesL[BATCH * 4 * NH];               // 64 x 32 f32 = 8 KB

  const int tid  = threadIdx.x;
  const int wave = tid >> 5, lane = tid & 31;
  const int lm = lane & 15, hf = lane >> 4;
  const int n0 = blockIdx.x * NH;

  // stage Whh slab: local row r -> global row (r/NH)*HID + n0 + (r%NH)
#if USE_ASYNC_LDS
  for (int idx = tid; idx < 4 * NH * (HID / 8); idx += 256){
    int r  = idx / (HID / 8);
    int ch = idx - r * (HID / 8);
    int grow = (r >> 3) * HID + n0 + (r & 7);
    __builtin_amdgcn_global_load_async_to_lds_b128(
        (gas1_v4i)(Whh + (size_t)grow * HID + ch * 8),
        (gas3_v4i)&slab[r * SLABSTR + ch * 8],
        0, 0);
  }
  __builtin_amdgcn_s_wait_asynccnt(0);
#else
  for (int idx = tid; idx < 4 * NH * (HID / 8); idx += 256){
    int r  = idx / (HID / 8);
    int ch = idx - r * (HID / 8);
    int grow = (r >> 3) * HID + n0 + (r & 7);
    *(uint4*)&slab[r * SLABSTR + ch * 8] =
        ((const uint4*)(Whh + (size_t)grow * HID))[ch];
  }
#endif
  __syncthreads();

  const int mt = wave & 3, nt = wave >> 2;               // 4 m-tiles x 2 n-tiles
  const int mrow = mt * 16 + lm;                         // batch row for A frags
  const u16* brow = &slab[(nt * 16 + lm) * SLABSTR];

  // cell ownership: 2 (batch, unit) elements per thread; c-state in registers
  const int e0 = tid, e1 = tid + 256;
  const int cb0 = e0 >> 3, cj0 = e0 & 7;
  const int cb1 = e1 >> 3, cj1 = e1 & 7;
  float cst0 = 0.f, cst1 = 0.f;

  // prefetch step-0 pre slice (64 rows x 4 gate-groups of 8 f32 = 256 x 32B)
  __builtin_prefetch(pre + (size_t)(tid >> 2) * FOURH + (tid & 3) * HID + n0, 0, 0);

  for (int t = 0; t < SEQT; ++t){
    const u16* hrd = hbuf + (size_t)(t & 1) * (BATCH * HID);
    u16*       hwr = hbuf + (size_t)((t + 1) & 1) * (BATCH * HID);
    const u16* arow = hrd + mrow * HID;

    // warm L2/L1 with next step's pre slice while this step computes
    if (t + 1 < SEQT)
      __builtin_prefetch(pre + ((size_t)(t + 1) * BATCH + (tid >> 2)) * FOURH
                             + (tid & 3) * HID + n0, 0, 0);

    v8f acc = (v8f){0.f,0.f,0.f,0.f,0.f,0.f,0.f,0.f};
    for (int k = 0; k < HID; k += 32){
      v16bf a = load_a_frag(arow, k, hf);                // global (L2-hot 96KB)
      v16bf b = load_b_frag(brow, k, hf);                // LDS
      acc = wmma_bf16(a, b, acc);
    }

    // epilogue: gates(LDS) = acc + pre[t]
    const float* prow = pre + (size_t)t * BATCH * FOURH;
    #pragma unroll
    for (int r = 0; r < 8; ++r){
      int m = mt * 16 + r + 8 * hf;
      int c = nt * 16 + lm;                              // local col = gate*8 + j
      int gcol = (c >> 3) * HID + n0 + (c & 7);
      gatesL[m * (4 * NH) + c] = acc[r] + prow[(size_t)m * FOURH + gcol];
    }
    __syncthreads();

    // LSTM cell (PyTorch gate order i,f,g,o)
    {
      const float* g = &gatesL[cb0 * (4 * NH)];
      float i_ = sigm(g[cj0]);
      float f_ = sigm(g[NH + cj0]);
      float gg = tanh_f(g[2 * NH + cj0]);
      float o_ = sigm(g[3 * NH + cj0]);
      cst0 = f_ * cst0 + i_ * gg;
      float h = o_ * tanh_f(cst0);
      hwr[cb0 * HID + n0 + cj0] = f2bf(h);
      size_t oidx = ((size_t)t * BATCH + cb0) * HID + n0 + cj0;
      float res = useRes ? bf2f(inseq[oidx]) : 0.f;
      outseq[oidx] = f2bf(h + res);
    }
    {
      const float* g = &gatesL[cb1 * (4 * NH)];
      float i_ = sigm(g[cj1]);
      float f_ = sigm(g[NH + cj1]);
      float gg = tanh_f(g[2 * NH + cj1]);
      float o_ = sigm(g[3 * NH + cj1]);
      cst1 = f_ * cst1 + i_ * gg;
      float h = o_ * tanh_f(cst1);
      hwr[cb1 * HID + n0 + cj1] = f2bf(h);
      size_t oidx = ((size_t)t * BATCH + cb1) * HID + n0 + cj1;
      float res = useRes ? bf2f(inseq[oidx]) : 0.f;
      outseq[oidx] = f2bf(h + res);
    }

    // device-wide step barrier (release h_{t+1}, acquire before next read)
    __builtin_amdgcn_fence(__ATOMIC_RELEASE, "agent");
    __syncthreads();
    if (tid == 0){
      atomicAdd(&cnt[t], 1u);
      while (__hip_atomic_load(&cnt[t], __ATOMIC_RELAXED, __HIP_MEMORY_SCOPE_AGENT)
             < (unsigned)gridDim.x)
        __builtin_amdgcn_s_sleep(1);
    }
    __syncthreads();
    __builtin_amdgcn_fence(__ATOMIC_ACQUIRE, "agent");
  }
}

// ---------- projection: out[b,t,o] = sum_k x3[t*B+b,k] * Wp[o,k] ----------
// Wp padded to [112 x 768] bf16; each wave: 16 rows x 7 n-tiles.
__global__ __launch_bounds__(256) void k_proj(const u16* __restrict__ A,
                                              const u16* __restrict__ Wp,
                                              float* __restrict__ out){
  const int tid  = threadIdx.x;
  const int wave = tid >> 5, lane = tid & 31;
  const int lm = lane & 15, hf = lane >> 4;
  const size_t mbase = (size_t)blockIdx.x * 128 + wave * 16;

  v8f acc[7];
  #pragma unroll
  for (int n = 0; n < 7; ++n) acc[n] = (v8f){0.f,0.f,0.f,0.f,0.f,0.f,0.f,0.f};

  const u16* arow = A + (mbase + lm) * HID;
  for (int k = 0; k < HID; k += 32){
    v16bf a = load_a_frag(arow, k, hf);
    #pragma unroll
    for (int n = 0; n < 7; ++n){
      v16bf b = load_b_frag(Wp + (size_t)(n * 16 + lm) * HID, k, hf);
      acc[n] = wmma_bf16(a, b, acc[n]);
    }
  }
  #pragma unroll
  for (int n = 0; n < 7; ++n)
    #pragma unroll
    for (int r = 0; r < 8; ++r){
      size_t m = mbase + r + 8 * hf;                     // m = t*B + b
      int    o = n * 16 + lm;
      if (o < OUTD){
        int t = (int)(m >> 6), b = (int)(m & 63);
        out[((size_t)b * SEQT + t) * OUTD + o] = acc[n][r];
      }
    }
}

// ---------- host launch ----------
extern "C" void kernel_launch(void* const* d_in, const int* in_sizes, int n_in,
                              void* d_out, int out_size, void* d_ws, size_t ws_size,
                              hipStream_t stream){
  (void)in_sizes; (void)n_in; (void)out_size; (void)ws_size;
  const float* x    = (const float*)d_in[0];
  const float* Wih1 = (const float*)d_in[1];
  const float* Whh1 = (const float*)d_in[2];
  const float* b1   = (const float*)d_in[3];
  const float* Wih2 = (const float*)d_in[4];
  const float* Whh2 = (const float*)d_in[5];
  const float* b2   = (const float*)d_in[6];
  const float* Wih3 = (const float*)d_in[7];
  const float* Whh3 = (const float*)d_in[8];
  const float* b3   = (const float*)d_in[9];
  const float* Wp   = (const float*)d_in[10];
  float* out = (float*)d_out;

  char* ws = (char*)d_ws;
  size_t off = 0;
  auto alloc = [&](size_t bytes)->char*{
    char* p = ws + off; off += (bytes + 255) & ~(size_t)255; return p;
  };
  u16*   xtb   = (u16*)  alloc((size_t)ROWS * INP * 2);      //  64 MB
  u16*   seq1  = (u16*)  alloc((size_t)ROWS * HID * 2);      //  96 MB
  u16*   seq2  = (u16*)  alloc((size_t)ROWS * HID * 2);
  u16*   seq3  = (u16*)  alloc((size_t)ROWS * HID * 2);
  float* pre   = (float*)alloc((size_t)ROWS * FOURH * 4);    // 768 MB
  u16*   wih1b = (u16*)  alloc((size_t)FOURH * INP * 2);
  u16*   wih2b = (u16*)  alloc((size_t)FOURH * HID * 2);
  u16*   wih3b = (u16*)  alloc((size_t)FOURH * HID * 2);
  u16*   whh1b = (u16*)  alloc((size_t)FOURH * HID * 2);
  u16*   whh2b = (u16*)  alloc((size_t)FOURH * HID * 2);
  u16*   whh3b = (u16*)  alloc((size_t)FOURH * HID * 2);
  u16*   wpb   = (u16*)  alloc((size_t)112 * HID * 2);
  u16*   hbuf  = (u16*)  alloc((size_t)2 * BATCH * HID * 2); // double-buffered h
  unsigned* cnt= (unsigned*)alloc((size_t)SEQT * 4);         // contiguous after hbuf

  const int zeroN = 2 * BATCH * HID / 2 + SEQT;              // hbuf dwords + cnt dwords

  // ---- precision conversion / packing ----
  k_cvt_x <<<(int)(((size_t)ROWS * INP) / 256), 256, 0, stream>>>(x, xtb);
  k_cvt   <<<(FOURH * INP + 255) / 256, 256, 0, stream>>>(Wih1, wih1b, FOURH * INP);
  k_cvt   <<<(FOURH * HID + 255) / 256, 256, 0, stream>>>(Wih2, wih2b, FOURH * HID);
  k_cvt   <<<(FOURH * HID + 255) / 256, 256, 0, stream>>>(Wih3, wih3b, FOURH * HID);
  k_cvt   <<<(FOURH * HID + 255) / 256, 256, 0, stream>>>(Whh1, whh1b, FOURH * HID);
  k_cvt   <<<(FOURH * HID + 255) / 256, 256, 0, stream>>>(Whh2, whh2b, FOURH * HID);
  k_cvt   <<<(FOURH * HID + 255) / 256, 256, 0, stream>>>(Whh3, whh3b, FOURH * HID);
  k_cvt_wp<<<(112 * HID) / 256, 256, 0, stream>>>(Wp, wpb);

  dim3 gGemm(ROWS / 64, FOURH / 256);

  // ---- layer 1 ----
  k_gemm_pre<<<gGemm, 256, 0, stream>>>(xtb, wih1b, b1, pre, INP);
  k_zero    <<<(zeroN + 255) / 256, 256, 0, stream>>>((unsigned*)hbuf, zeroN);
  k_scan    <<<SCAN_WGS, 256, 0, stream>>>(pre, whh1b, nullptr, seq1, hbuf, cnt, 0);

  // ---- layer 2 (residual) ----
  k_gemm_pre<<<gGemm, 256, 0, stream>>>(seq1, wih2b, b2, pre, HID);
  k_zero    <<<(zeroN + 255) / 256, 256, 0, stream>>>((unsigned*)hbuf, zeroN);
  k_scan    <<<SCAN_WGS, 256, 0, stream>>>(pre, whh2b, seq1, seq2, hbuf, cnt, 1);

  // ---- layer 3 (residual) ----
  k_gemm_pre<<<gGemm, 256, 0, stream>>>(seq2, wih3b, b3, pre, HID);
  k_zero    <<<(zeroN + 255) / 256, 256, 0, stream>>>((unsigned*)hbuf, zeroN);
  k_scan    <<<SCAN_WGS, 256, 0, stream>>>(pre, whh3b, seq2, seq3, hbuf, cnt, 1);

  // ---- projection ----
  k_proj<<<ROWS / 128, 256, 0, stream>>>(seq3, wpb, out);
}
